// SignalGraphAttention_43937515438699
// MI455X (gfx1250) — matrix operand order repaired and checked by reading, not compiled, not run on hardware
//
#include <hip/hip_runtime.h>
#include <hip/hip_bf16.h>

// ---------------------------------------------------------------------------
// SignalGraphAttention for MI455X (gfx1250, wave32, WMMA + async-to-LDS).
// All matrix math goes through v_wmma_f32_16x16x32_f16 (K=32 == D_HEAD).
// ---------------------------------------------------------------------------

#define DMODEL 256
#define NHEADS 8
#define DHEAD  32
#define BATCH  8
#define NNODE  1024
#define NEDGE  16384
#define NTOK   (BATCH * NNODE)      // 8192
#define NEGV   (-1.0e30f)

typedef __attribute__((ext_vector_type(16))) _Float16 v16h;
typedef __attribute__((ext_vector_type(8)))  float    v8f;

// ---- gfx1250 async global->LDS copy (ASYNCcnt-tracked) ---------------------
// ISA 10.2: flat addresses in the LDS aperture carry the LDS byte offset in
// bits [31:0], so truncating a generic pointer to a __shared__ object yields
// exactly the VDST operand the async instruction wants (HW adds LDS_BASE).
static __device__ inline void async_load_b128(const void* lds_dst, const void* gsrc) {
  const unsigned           l = (unsigned)(unsigned long long)lds_dst;
  const unsigned long long g = (unsigned long long)gsrc;
  asm volatile("global_load_async_to_lds_b128 %0, %1, off"
               :: "v"(l), "v"(g) : "memory");
}
static __device__ inline void wait_async_zero() {
  asm volatile("s_wait_asynccnt 0x0" ::: "memory");
}

// ---- WMMA fragment helpers (contiguous k = 16*(lane/16) + s, wave32) --------------

// A-matrix 16x32 f16, element (m,k), row-major source with leading dim ld.
static __device__ inline v16h load_a_frag(const _Float16* __restrict__ base, int ld) {
  const int lane = threadIdx.x & 31;
  const int row  = lane & 15;
  const int koff = (lane >> 4) << 4;   // 0 or 16
  const _Float16* p = base + (size_t)row * ld + koff;
  v16h a;
#pragma unroll
  for (int i = 0; i < 8; ++i) a[i] = p[i];
#pragma unroll
  for (int i = 0; i < 8; ++i) a[8 + i] = p[8 + i];
  return a;
}

// B-matrix 32x16 f16, element (k,n) = base[n*ld + k] (k contiguous per lane).
static __device__ inline v16h load_b_frag_kcontig(const _Float16* __restrict__ base, int ld) {
  const int lane = threadIdx.x & 31;
  const _Float16* p = base + (size_t)(lane & 15) * ld + ((lane >> 4) << 4);
  v16h b;
#pragma unroll
  for (int h = 0; h < 16; ++h) b[h] = p[h];
  return b;
}

static __device__ inline v8f wmma_f32_16x16x32(v16h a, v16h b, v8f c) {
  return __builtin_amdgcn_wmma_f32_16x16x32_f16(false, a, false, b, (short)0, c,
                                                false, false);
}

// ---------------------------------------------------------------------------
// 1) LayerNorm: one wave per 256-wide row, f32 statistics, f16 output.
// ---------------------------------------------------------------------------
__global__ __launch_bounds__(256) void ln_kernel(const float* __restrict__ x,
                                                 const float* __restrict__ gamma,
                                                 const float* __restrict__ beta,
                                                 _Float16* __restrict__ xn) {
  const int wave = threadIdx.x >> 5, lane = threadIdx.x & 31;
  const int row = blockIdx.x * 8 + wave;
  const float* xr = x + (size_t)row * DMODEL;
  float v[8], s = 0.f, ss = 0.f;
#pragma unroll
  for (int i = 0; i < 8; ++i) { v[i] = xr[lane + 32 * i]; s += v[i]; ss += v[i] * v[i]; }
#pragma unroll
  for (int off = 16; off > 0; off >>= 1) {
    s  += __shfl_xor(s,  off, 32);
    ss += __shfl_xor(ss, off, 32);
  }
  const float mu = s * (1.f / DMODEL);
  const float var = ss * (1.f / DMODEL) - mu * mu;
  const float r = rsqrtf(var + 1e-5f);
  _Float16* o = xn + (size_t)row * DMODEL;
#pragma unroll
  for (int i = 0; i < 8; ++i) {
    const int c = lane + 32 * i;
    o[c] = (_Float16)((v[i] - mu) * r * gamma[c] + beta[c]);
  }
}

// ---------------------------------------------------------------------------
// 2) Pack Wq|Wk|Wv|Wo into one f16 buffer (weights are L2-resident, 0.5 MB).
// ---------------------------------------------------------------------------
__global__ __launch_bounds__(256) void cvt_weights_kernel(const float* __restrict__ Wq,
                                                          const float* __restrict__ Wk,
                                                          const float* __restrict__ Wv,
                                                          const float* __restrict__ Wo,
                                                          _Float16* __restrict__ W4) {
  const int i = blockIdx.x * 256 + threadIdx.x;          // 0 .. 4*65536-1
  const int sel = i >> 16;
  const float* src = (sel == 0) ? Wq : (sel == 1) ? Wk : (sel == 2) ? Wv : Wo;
  W4[i] = (_Float16)src[i & 0xFFFF];
}

// ---------------------------------------------------------------------------
// 3) QKV projections: D = xn @ W^T + b.  Wave computes 16x64 strip (32 WMMAs).
//    Output written head-major [b][h][n][d] (f16) for attention fragment loads.
// ---------------------------------------------------------------------------
__global__ __launch_bounds__(256) void qkv_kernel(const _Float16* __restrict__ xn,
                                                  const _Float16* __restrict__ W4,
                                                  const float* __restrict__ bq,
                                                  const float* __restrict__ bk,
                                                  const float* __restrict__ bv,
                                                  _Float16* __restrict__ Qh,
                                                  _Float16* __restrict__ Kh,
                                                  _Float16* __restrict__ Vh) {
  const int wave = threadIdx.x >> 5, lane = threadIdx.x & 31;
  const int row0 = blockIdx.x * 128 + wave * 16;
  const int col0 = blockIdx.y * 64;
  const int which = blockIdx.z;
  const _Float16* W = W4 + (size_t)which * DMODEL * DMODEL;  // W[col][k]
  const float* bias = (which == 0) ? bq : (which == 1) ? bk : bv;
  _Float16* dst = (which == 0) ? Qh : (which == 1) ? Kh : Vh;

  v8f a0 = {}, a1 = {}, a2 = {}, a3 = {};
#pragma unroll
  for (int k0 = 0; k0 < DMODEL; k0 += 32) {
    if (k0 + 32 < DMODEL)
      __builtin_prefetch(xn + (size_t)row0 * DMODEL + k0 + 32, 0, 1); // global_prefetch_b8
    const v16h a = load_a_frag(xn + (size_t)row0 * DMODEL + k0, DMODEL);
    a0 = wmma_f32_16x16x32(a, load_b_frag_kcontig(W + (size_t)(col0 +  0) * DMODEL + k0, DMODEL), a0);
    a1 = wmma_f32_16x16x32(a, load_b_frag_kcontig(W + (size_t)(col0 + 16) * DMODEL + k0, DMODEL), a1);
    a2 = wmma_f32_16x16x32(a, load_b_frag_kcontig(W + (size_t)(col0 + 32) * DMODEL + k0, DMODEL), a2);
    a3 = wmma_f32_16x16x32(a, load_b_frag_kcontig(W + (size_t)(col0 + 48) * DMODEL + k0, DMODEL), a3);
  }
  const int rbase = (lane >> 4) << 3;   // C/D layout: m = r + 8*(lane>=16)
#pragma unroll
  for (int r = 0; r < 8; ++r) {
    const int token = row0 + rbase + r;
    const int bb = token >> 10, nn = token & 1023;
#pragma unroll
    for (int t = 0; t < 4; ++t) {
      const int col = col0 + t * 16 + (lane & 15);
      const float acc = (t == 0) ? a0[r] : (t == 1) ? a1[r] : (t == 2) ? a2[r] : a3[r];
      const int hh = col >> 5, dd = col & 31;
      dst[((size_t)(bb * NHEADS + hh) * NNODE + nn) * DHEAD + dd] =
          (_Float16)(acc + bias[col]);
    }
  }
}

// ---------------------------------------------------------------------------
// 4) Mask bitmap: bit(n,m) set iff edge src=n,tgt=m exists, or n==m.
// ---------------------------------------------------------------------------
__global__ __launch_bounds__(256) void mask_init_kernel(unsigned* __restrict__ maskw) {
  const int w = blockIdx.x * 256 + threadIdx.x;          // 0 .. 1024*32-1
  const int row = w >> 5, wcol = w & 31;
  maskw[w] = (wcol == (row >> 5)) ? (1u << (row & 31)) : 0u;  // diagonal
}

__global__ __launch_bounds__(256) void mask_edges_kernel(const int* __restrict__ ei,
                                                         unsigned* __restrict__ maskw) {
  const int e = blockIdx.x * 256 + threadIdx.x;
  const int s = ei[e], t = ei[NEDGE + e];
  atomicOr(&maskw[(size_t)s * 32 + (t >> 5)], 1u << (t & 31));
}

// ---------------------------------------------------------------------------
// 5) CSR of edges keyed by tgt (bias is added at attn[:, :, tgt, src]).
// ---------------------------------------------------------------------------
__global__ __launch_bounds__(256) void zero_i32(int* __restrict__ p, int n) {
  const int i = blockIdx.x * 256 + threadIdx.x;
  if (i < n) p[i] = 0;
}

__global__ __launch_bounds__(256) void csr_count_kernel(const int* __restrict__ ei,
                                                        int* __restrict__ counts) {
  const int e = blockIdx.x * 256 + threadIdx.x;
  atomicAdd(&counts[ei[NEDGE + e]], 1);
}

__global__ __launch_bounds__(1024) void csr_scan_kernel(const int* __restrict__ counts,
                                                        int* __restrict__ starts) {
  __shared__ int tmp[NNODE];
  const int t = threadIdx.x;
  tmp[t] = counts[t];
  __syncthreads();
  for (int off = 1; off < NNODE; off <<= 1) {
    const int v = (t >= off) ? tmp[t - off] : 0;
    __syncthreads();
    tmp[t] += v;
    __syncthreads();
  }
  starts[t + 1] = tmp[t];
  if (t == 0) starts[0] = 0;
}

__global__ __launch_bounds__(256) void csr_fill_kernel(const int* __restrict__ ei,
                                                       const int* __restrict__ et,
                                                       const int* __restrict__ starts,
                                                       int* __restrict__ cursor,
                                                       int* __restrict__ csr_src,
                                                       int* __restrict__ csr_et) {
  const int e = blockIdx.x * 256 + threadIdx.x;
  const int t = ei[NEDGE + e];
  const int pos = starts[t] + atomicAdd(&cursor[t], 1);
  csr_src[pos] = ei[e];
  csr_et[pos]  = et[e];
}

// ---------------------------------------------------------------------------
// 6) Attention. 8 waves/block share one (b,h); each wave owns 16 query rows.
//    Per 64-key tile (block-cooperative):
//      stage K via global_load_async_to_lds_b128 (ASYNCcnt) and V transposed
//      into LDS -> 4 score WMMAs (B-frags from LDS, contiguous ds_load_b128)
//      -> scale/mask -> CSR edge-bias scatter (ds_add_f32) -> online softmax
//      (wave-private LDS stats, per-wave LDS ordering, no extra barriers)
//      -> P cast to f16 in LDS -> 4 PV WMMAs (B-frags from transposed V).
// ---------------------------------------------------------------------------
__global__ __launch_bounds__(256) void attn_kernel(const _Float16* __restrict__ Qh,
                                                   const _Float16* __restrict__ Kh,
                                                   const _Float16* __restrict__ Vh,
                                                   const unsigned* __restrict__ maskw,
                                                   const int* __restrict__ starts,
                                                   const int* __restrict__ csr_src,
                                                   const int* __restrict__ csr_et,
                                                   const float* __restrict__ eb,
                                                   _Float16* __restrict__ Oh) {
  __shared__ float    sS[8][16][64];      // 32 KB score tiles (f32, per wave)
  __shared__ _Float16 sP[8][16][64];      // 16 KB probabilities (f16, per wave)
  __shared__ _Float16 sK[64][DHEAD];      //  4 KB key tile   (key, d), shared
  __shared__ _Float16 sVt[DHEAD][64];     //  4 KB value tile (d, key), shared
  __shared__ float    sM[8][16], sL[8][16], sA[8][16];

  const int wave = threadIdx.x >> 5;
  const int lane = threadIdx.x & 31;
  const int bh = blockIdx.y;
  const int b = bh >> 3, h = bh & 7;
  const int row0 = blockIdx.x * 128 + wave * 16;
  const int rhalf = lane >> 4;            // 0 or 1
  const int rbase = rhalf << 3;           // C/D row base

  const _Float16* Qb = Qh + (size_t)bh * NNODE * DHEAD;
  const _Float16* Kb = Kh + (size_t)bh * NNODE * DHEAD;
  const _Float16* Vb = Vh + (size_t)bh * NNODE * DHEAD;

  if (lane < 16) { sM[wave][lane] = NEGV; sL[wave][lane] = 0.f; }

  // staging decomposition: 256 threads x 16B cover one 64x32 f16 tile
  const int skey   = threadIdx.x >> 2;        // 0..63
  const int schunk = threadIdx.x & 3;         // 16B chunk within 64B row

  const v16h qa = load_a_frag(Qb + (size_t)row0 * DHEAD, DHEAD);
  v8f o0 = {}, o1 = {};
  const float scale = 0.17677669529663687f;   // 1/sqrt(32)

  for (int m0 = 0; m0 < NNODE; m0 += 64) {
    __syncthreads();   // previous tile fully consumed before restaging

    // ---- K tile: async DMA global -> LDS (16 B per thread) ----------------
    async_load_b128(&sK[skey][schunk * 8],
                    Kb + (size_t)(m0 + skey) * DHEAD + schunk * 8);
    // ---- V tile: vector load + transposed LDS store -----------------------
    {
      const uint4 vv = *(const uint4*)(Vb + (size_t)(m0 + skey) * DHEAD + schunk * 8);
      const _Float16* hv = (const _Float16*)&vv;
#pragma unroll
      for (int j = 0; j < 8; ++j) sVt[schunk * 8 + j][skey] = hv[j];
    }
    wait_async_zero();
    __syncthreads();   // K/V tile visible to all waves

    // ---- scores: S(16x64) = Q(16x32) @ K^T, B-frags from LDS --------------
    v8f z = {};
    v8f s0 = wmma_f32_16x16x32(qa, load_b_frag_kcontig(&sK[ 0][0], DHEAD), z);
    v8f s1 = wmma_f32_16x16x32(qa, load_b_frag_kcontig(&sK[16][0], DHEAD), z);
    v8f s2 = wmma_f32_16x16x32(qa, load_b_frag_kcontig(&sK[32][0], DHEAD), z);
    v8f s3 = wmma_f32_16x16x32(qa, load_b_frag_kcontig(&sK[48][0], DHEAD), z);

    // ---- scale + mask, spill to wave-private LDS tile ---------------------
#pragma unroll
    for (int t = 0; t < 4; ++t) {
      const int colf = t * 16 + (lane & 15);
      const int m = m0 + colf;
#pragma unroll
      for (int r = 0; r < 8; ++r) {
        const int rowf = rbase + r;
        float val = ((t == 0) ? s0[r] : (t == 1) ? s1[r] : (t == 2) ? s2[r] : s3[r]) * scale;
        const unsigned mw = maskw[(size_t)(row0 + rowf) * 32 + (m >> 5)];
        if (!((mw >> (m & 31)) & 1u)) val = NEGV;
        sS[wave][rowf][colf] = val;
      }
    }

    // ---- edge bias: rows keyed by tgt, scatter to col=src in this tile ----
    for (int rr = 0; rr < 16; ++rr) {
      const int qrow = row0 + rr;
      const int e0 = starts[qrow], e1 = starts[qrow + 1];
      const _Float16* qr = Qb + (size_t)qrow * DHEAD;
      for (int e = e0 + lane; e < e1; e += 32) {
        const int srcc = csr_src[e];
        if (srcc >= m0 && srcc < m0 + 64) {
          const float* ebr = eb + csr_et[e] * DHEAD;
          float dot = 0.f;
#pragma unroll
          for (int d = 0; d < DHEAD; ++d) dot += (float)qr[d] * ebr[d];
          atomicAdd(&sS[wave][rr][srcc - m0], dot);   // ds_add_f32
        }
      }
    }

    // ---- online softmax update: lanes (l, l^16) cooperate on row l&15 -----
    {
      const int rowf = lane & 15;
      const int c0 = rhalf * 32;
      float tmax = NEGV;
#pragma unroll
      for (int c = 0; c < 32; ++c) tmax = fmaxf(tmax, sS[wave][rowf][c0 + c]);
      const float rowmax = fmaxf(tmax, __shfl_xor(tmax, 16, 32));
      const float mold = sM[wave][rowf];
      const float mnew = fmaxf(mold, rowmax);
      const float alpha = (mold == mnew) ? 1.f : __expf(mold - mnew);
      float psum = 0.f;
#pragma unroll
      for (int c = 0; c < 32; ++c) {
        const float sv = sS[wave][rowf][c0 + c];
        const float p = (sv > 0.5f * NEGV) ? __expf(sv - mnew) : 0.f;
        sP[wave][rowf][c0 + c] = (_Float16)p;
        psum += p;
      }
      psum += __shfl_xor(psum, 16, 32);
      sM[wave][rowf] = mnew;
      sL[wave][rowf] = alpha * sL[wave][rowf] + psum;
      sA[wave][rowf] = alpha;
    }

    // ---- rescale accumulators, then O += P(16x64) @ V(64x32) --------------
#pragma unroll
    for (int r = 0; r < 8; ++r) {
      const float a = sA[wave][rbase + r];
      o0[r] *= a; o1[r] *= a;
    }
    const v16h pa0 = load_a_frag(&sP[wave][0][0],  64);   // P cols 0..31
    const v16h pa1 = load_a_frag(&sP[wave][0][32], 64);   // P cols 32..63
    o0 = wmma_f32_16x16x32(pa0, load_b_frag_kcontig(&sVt[ 0][ 0], 64), o0);
    o0 = wmma_f32_16x16x32(pa1, load_b_frag_kcontig(&sVt[ 0][32], 64), o0);
    o1 = wmma_f32_16x16x32(pa0, load_b_frag_kcontig(&sVt[16][ 0], 64), o1);
    o1 = wmma_f32_16x16x32(pa1, load_b_frag_kcontig(&sVt[16][32], 64), o1);
  }

  // ---- normalize and write O back in model layout [b][n][h*32+d] (f16) ----
#pragma unroll
  for (int r = 0; r < 8; ++r) {
    const int rowf = rbase + r;
    const float inv = 1.f / sL[wave][rowf];
    const size_t base = ((size_t)(b * NNODE + row0 + rowf)) * DMODEL + h * DHEAD + (lane & 15);
    Oh[base]      = (_Float16)(o0[r] * inv);
    Oh[base + 16] = (_Float16)(o1[r] * inv);
  }
}

// ---------------------------------------------------------------------------
// 7) Output projection + bias + residual: out = x + O @ Wo^T + bo  (f32 out)
// ---------------------------------------------------------------------------
__global__ __launch_bounds__(256) void out_proj_kernel(const _Float16* __restrict__ Oh,
                                                       const _Float16* __restrict__ Wo,
                                                       const float* __restrict__ bo,
                                                       const float* __restrict__ x,
                                                       float* __restrict__ out) {
  const int wave = threadIdx.x >> 5, lane = threadIdx.x & 31;
  const int row0 = blockIdx.x * 128 + wave * 16;
  const int col0 = blockIdx.y * 64;

  v8f a0 = {}, a1 = {}, a2 = {}, a3 = {};
#pragma unroll
  for (int k0 = 0; k0 < DMODEL; k0 += 32) {
    const v16h a = load_a_frag(Oh + (size_t)row0 * DMODEL + k0, DMODEL);
    a0 = wmma_f32_16x16x32(a, load_b_frag_kcontig(Wo + (size_t)(col0 +  0) * DMODEL + k0, DMODEL), a0);
    a1 = wmma_f32_16x16x32(a, load_b_frag_kcontig(Wo + (size_t)(col0 + 16) * DMODEL + k0, DMODEL), a1);
    a2 = wmma_f32_16x16x32(a, load_b_frag_kcontig(Wo + (size_t)(col0 + 32) * DMODEL + k0, DMODEL), a2);
    a3 = wmma_f32_16x16x32(a, load_b_frag_kcontig(Wo + (size_t)(col0 + 48) * DMODEL + k0, DMODEL), a3);
  }
  const int rbase = (lane >> 4) << 3;
#pragma unroll
  for (int r = 0; r < 8; ++r) {
    const size_t token = (size_t)(row0 + rbase + r);
#pragma unroll
    for (int t = 0; t < 4; ++t) {
      const int col = col0 + t * 16 + (lane & 15);
      const float acc = (t == 0) ? a0[r] : (t == 1) ? a1[r] : (t == 2) ? a2[r] : a3[r];
      out[token * DMODEL + col] = acc + bo[col] + x[token * DMODEL + col];
    }
  }
}

// ---------------------------------------------------------------------------
extern "C" void kernel_launch(void* const* d_in, const int* in_sizes, int n_in,
                              void* d_out, int out_size, void* d_ws, size_t ws_size,
                              hipStream_t stream) {
  (void)in_sizes; (void)n_in; (void)out_size; (void)ws_size;
  const float* x  = (const float*)d_in[0];
  const int*   ei = (const int*)d_in[1];
  const int*   et = (const int*)d_in[2];
  const float* Wq = (const float*)d_in[3];
  const float* bq = (const float*)d_in[4];
  const float* Wk = (const float*)d_in[5];
  const float* bk = (const float*)d_in[6];
  const float* Wv = (const float*)d_in[7];
  const float* bv = (const float*)d_in[8];
  const float* Wo = (const float*)d_in[9];
  const float* bo = (const float*)d_in[10];
  const float* eb = (const float*)d_in[11];
  const float* lg = (const float*)d_in[12];
  const float* lb = (const float*)d_in[13];
  float* out = (float*)d_out;

  // Workspace carve-out (~21 MB), 256 B aligned slices.
  char* w = (char*)d_ws;
  size_t off = 0;
  auto take = [&](size_t bytes) -> void* {
    void* p = w + off;
    off += (bytes + 255) & ~(size_t)255;
    return p;
  };
  _Float16* xn_h = (_Float16*)take((size_t)NTOK * DMODEL * 2);
  _Float16* W4_h = (_Float16*)take((size_t)4 * DMODEL * DMODEL * 2);
  _Float16* Qh   = (_Float16*)take((size_t)NTOK * DMODEL * 2);
  _Float16* Kh   = (_Float16*)take((size_t)NTOK * DMODEL * 2);
  _Float16* Vh   = (_Float16*)take((size_t)NTOK * DMODEL * 2);
  _Float16* Oh   = (_Float16*)take((size_t)NTOK * DMODEL * 2);
  unsigned* maskw  = (unsigned*)take((size_t)NNODE * 32 * 4);
  int* counts  = (int*)take((size_t)NNODE * 4);
  int* cursor  = (int*)take((size_t)NNODE * 4);
  int* starts  = (int*)take((size_t)(NNODE + 1) * 4);
  int* csr_src = (int*)take((size_t)NEDGE * 4);
  int* csr_et  = (int*)take((size_t)NEDGE * 4);

  // 1) LayerNorm -> f16
  ln_kernel<<<NTOK / 8, 256, 0, stream>>>(x, lg, lb, xn_h);
  // 2) weights -> f16
  cvt_weights_kernel<<<(4 * DMODEL * DMODEL) / 256, 256, 0, stream>>>(Wq, Wk, Wv, Wo, W4_h);
  // 3) Q/K/V projections (WMMA)
  qkv_kernel<<<dim3(NTOK / 128, DMODEL / 64, 3), 256, 0, stream>>>(
      xn_h, W4_h, bq, bk, bv, Qh, Kh, Vh);
  // 4) mask bitmap
  mask_init_kernel<<<(NNODE * 32) / 256, 256, 0, stream>>>(maskw);
  mask_edges_kernel<<<NEDGE / 256, 256, 0, stream>>>(ei, maskw);
  // 5) CSR by tgt
  zero_i32<<<(NNODE + 255) / 256, 256, 0, stream>>>(counts, NNODE);
  zero_i32<<<(NNODE + 255) / 256, 256, 0, stream>>>(cursor, NNODE);
  csr_count_kernel<<<NEDGE / 256, 256, 0, stream>>>(ei, counts);
  csr_scan_kernel<<<1, 1024, 0, stream>>>(counts, starts);
  csr_fill_kernel<<<NEDGE / 256, 256, 0, stream>>>(ei, et, starts, cursor, csr_src, csr_et);
  // 6) masked + biased flash attention (WMMA + async-to-LDS staging)
  attn_kernel<<<dim3(NNODE / 128, BATCH * NHEADS), 256, 0, stream>>>(
      Qh, Kh, Vh, maskw, starts, csr_src, csr_et, eb, Oh);
  // 7) output projection + residual (WMMA)
  out_proj_kernel<<<dim3(NTOK / 128, DMODEL / 64), 256, 0, stream>>>(Oh, W4_h + 3 * DMODEL * DMODEL, bo, x, out);
}
